// MultilayerGRU_67671504715964
// MI455X (gfx1250) — matrix operand order repaired and checked by reading, not compiled, not run on hardware
//
#include <hip/hip_runtime.h>

#define GB 32      // batch
#define GS 4096    // sequence length
#define GH 256     // hidden size
#define GO 256     // output size
#define GL 2       // layers
#define HH (GH*GH)

typedef __bf16 v16bf __attribute__((ext_vector_type(16)));
typedef float  v8f   __attribute__((ext_vector_type(8)));
typedef float  v4f   __attribute__((ext_vector_type(4)));
typedef int    v4i   __attribute__((ext_vector_type(4)));

union BF { v16bf v; v4i q[2]; };
union FV { v8f v; v4f q[2]; };

__device__ __forceinline__ __bf16 f2bf(float f) {
  unsigned u = __builtin_bit_cast(unsigned, f);
  u += 0x7FFFu + ((u >> 16) & 1u);                 // round-to-nearest-even
  unsigned short s = (unsigned short)(u >> 16);
  return __builtin_bit_cast(__bf16, s);
}

__device__ __forceinline__ float sigmoidf_(float x) {
  return 1.0f / (1.0f + __expf(-x));
}
__device__ __forceinline__ float tanhf_(float x) {
  float ax = fabsf(x);
  float e  = __expf(-2.0f * ax);
  float t  = (1.0f - e) / (1.0f + e);
  return x < 0.0f ? -t : t;
}

__device__ __forceinline__ v8f wmma_bf16(BF a, BF b, v8f c) {
  // D(16x16,f32) = A(16x32,bf16) x B(32x16,bf16) + C
  return __builtin_amdgcn_wmma_f32_16x16x32_bf16(false, a.v, false, b.v,
                                                 (short)0, c, false, false);
}

// ---------------------------------------------------------------------------
// f32 -> bf16 conversion (weights)
// ---------------------------------------------------------------------------
__global__ __launch_bounds__(256) void cvt_bf16(const float* __restrict__ s,
                                                __bf16* __restrict__ d, int n) {
  int i = blockIdx.x * 256 + threadIdx.x;
  if (i < n) d[i] = f2bf(s[i]);
}

// ---------------------------------------------------------------------------
// X-projection GEMM: for one layer computes, for all t and all 3 gates,
//   Xp frag(t,gate,mt,nt)[lane*8 + i] = bias + A[b,:] @ W[gate][n,:]^T
// (fragment stored lane-major so both producer stores and consumer loads are
//  contiguous 32B per lane -> b128 traffic on both sides)
// A is either f32 (network input) or bf16 (previous layer output), [B,S,H].
// grid = (2 mtiles, S), block = 256 (8 waves); each wave: 2 ntiles x 3 gates.
// ---------------------------------------------------------------------------
__global__ __launch_bounds__(256) void xproj_gemm(
    const float* __restrict__ xf, const __bf16* __restrict__ xb, int a_bf16,
    const __bf16* __restrict__ Wz, const __bf16* __restrict__ Wr,
    const __bf16* __restrict__ Wg,
    const float* __restrict__ bz, const float* __restrict__ br,
    const float* __restrict__ bg,
    float* __restrict__ Xp) {
  __shared__ __bf16 As[16][GH + 8];
  const int t = blockIdx.y, mt = blockIdx.x, tid = threadIdx.x;
  {
    const int row = tid >> 4, c0 = (tid & 15) * 16;
    const size_t base = ((size_t)(mt * 16 + row) * GS + t) * GH + c0;
    if (a_bf16) {
      const __bf16* src = xb + base;
      *(v4i*)&As[row][c0]     = *(const v4i*)src;
      *(v4i*)&As[row][c0 + 8] = *(const v4i*)(src + 8);
    } else {
      const v4f* src = (const v4f*)(xf + base);
      v4f f0 = src[0], f1 = src[1], f2 = src[2], f3 = src[3];
      union { __bf16 h[8]; v4i q; } p0, p1;
      for (int e = 0; e < 4; ++e) { p0.h[e] = f2bf(f0[e]); p0.h[4 + e] = f2bf(f1[e]); }
      for (int e = 0; e < 4; ++e) { p1.h[e] = f2bf(f2[e]); p1.h[4 + e] = f2bf(f3[e]); }
      *(v4i*)&As[row][c0]     = p0.q;
      *(v4i*)&As[row][c0 + 8] = p1.q;
    }
  }
  __syncthreads();
  const int w = tid >> 5, l = tid & 31, lr = l & 15, lh = l >> 4;
  for (int jj = 0; jj < 2; ++jj) {
    const int nt = w * 2 + jj;
    const int n  = nt * 16 + lr;
    FV az, ar, ag;
    { float v = bz[n]; for (int i = 0; i < 8; ++i) az.v[i] = v; }
    { float v = br[n]; for (int i = 0; i < 8; ++i) ar.v[i] = v; }
    { float v = bg[n]; for (int i = 0; i < 8; ++i) ag.v[i] = v; }
    for (int ks = 0; ks < 8; ++ks) {
      BF a, b;
      const __bf16* ap = &As[lr][ks * 32 + lh * 16];
      a.q[0] = *(const v4i*)ap;
      a.q[1] = *(const v4i*)(ap + 8);
      const size_t koff = (size_t)n * GH + ks * 32 + lh * 16;
      b.q[0] = *(const v4i*)(Wz + koff);
      b.q[1] = *(const v4i*)(Wz + koff + 8);
      az.v = wmma_bf16(a, b, az.v);
      b.q[0] = *(const v4i*)(Wr + koff);
      b.q[1] = *(const v4i*)(Wr + koff + 8);
      ar.v = wmma_bf16(a, b, ar.v);
      b.q[0] = *(const v4i*)(Wg + koff);
      b.q[1] = *(const v4i*)(Wg + koff + 8);
      ag.v = wmma_bf16(a, b, ag.v);
    }
    // lane-major fragment: contiguous 32B per lane -> b128 stores
    float* dz = Xp + (((size_t)(t * 3 + 0) * 2 + mt) * 16 + nt) * 256 + l * 8;
    float* dr = Xp + (((size_t)(t * 3 + 1) * 2 + mt) * 16 + nt) * 256 + l * 8;
    float* dg = Xp + (((size_t)(t * 3 + 2) * 2 + mt) * 16 + nt) * 256 + l * 8;
    *(v4f*)dz = az.q[0]; *(v4f*)(dz + 4) = az.q[1];
    *(v4f*)dr = ar.q[0]; *(v4f*)(dr + 4) = ar.q[1];
    *(v4f*)dg = ag.q[0]; *(v4f*)(dg + 4) = ag.q[1];
  }
}

// ---------------------------------------------------------------------------
// Sequential GRU scan for one layer. Single workgroup, 1024 threads = 32
// waves; wave w owns output tile (mt = w&1, nt = w>>1) of the [32,256] state.
// h kept in LDS (bf16, WMMA A-side) + owner registers (f32). Recurrent
// weights stream from L2 as B-fragments every step. 2 barriers / step.
// ---------------------------------------------------------------------------
__global__ __launch_bounds__(1024) void gru_scan(
    const float* __restrict__ Xp,
    const __bf16* __restrict__ Wz, const __bf16* __restrict__ Wr,
    const __bf16* __restrict__ Wg,
    const float* __restrict__ h0,   // [B,L,H]
    const int layer,
    __bf16* __restrict__ outb,      // [B,S,H] layer output (bf16)
    float* __restrict__ hout) {     // [B,L,H] final-h region of d_out
  __shared__ __bf16 hbf [GB][GH + 8];
  __shared__ __bf16 hrbf[GB][GH + 8];
  const int tid = threadIdx.x;
  const int w = tid >> 5, l = tid & 31, lr = l & 15, lh = l >> 4;
  const int mt = w & 1, nt = w >> 1;
  const int N = nt * 16 + lr;

  for (int e = 0; e < 8; ++e) {
    int idx = tid * 8 + e, b = idx >> 8, n = idx & 255;
    hbf[b][n] = f2bf(h0[((size_t)b * GL + layer) * GH + n]);
  }
  float hreg[8];
  for (int i = 0; i < 8; ++i) {
    int M = mt * 16 + lh * 8 + i;
    hreg[i] = h0[((size_t)M * GL + layer) * GH + N];
  }
  __syncthreads();

  // Xp fragment base for this wave at t=0; step stride = 3*2*16*256 floats.
  const float* xp = Xp + (((size_t)mt * 16 + nt) * 256) + l * 8;
  for (int t = 0; t < GS; ++t, xp += 3 * 2 * 16 * 256) {
    FV az, ar;
    az.q[0] = *(const v4f*)xp;            az.q[1] = *(const v4f*)(xp + 4);
    ar.q[0] = *(const v4f*)(xp + 8192);   ar.q[1] = *(const v4f*)(xp + 8196);
    for (int ks = 0; ks < 8; ++ks) {
      BF a, b;
      const __bf16* ap = &hbf[mt * 16 + lr][ks * 32 + lh * 16];
      a.q[0] = *(const v4i*)ap;
      a.q[1] = *(const v4i*)(ap + 8);
      const size_t koff = (size_t)N * GH + ks * 32 + lh * 16;
      b.q[0] = *(const v4i*)(Wz + koff);
      b.q[1] = *(const v4i*)(Wz + koff + 8);
      az.v = wmma_bf16(a, b, az.v);
      b.q[0] = *(const v4i*)(Wr + koff);
      b.q[1] = *(const v4i*)(Wr + koff + 8);
      ar.v = wmma_bf16(a, b, ar.v);
    }
    float zv[8];
    for (int i = 0; i < 8; ++i) {
      zv[i]   = sigmoidf_(az.v[i]);
      float r = sigmoidf_(ar.v[i]);
      int M = mt * 16 + lh * 8 + i;
      hrbf[M][N] = f2bf(hreg[i] * r);       // (h ∘ r) as next A-matrix
    }
    __syncthreads();
    FV ag;
    ag.q[0] = *(const v4f*)(xp + 16384);  ag.q[1] = *(const v4f*)(xp + 16388);
    for (int ks = 0; ks < 8; ++ks) {
      BF a, b;
      const __bf16* ap = &hrbf[mt * 16 + lr][ks * 32 + lh * 16];
      a.q[0] = *(const v4i*)ap;
      a.q[1] = *(const v4i*)(ap + 8);
      const size_t koff = (size_t)N * GH + ks * 32 + lh * 16;
      b.q[0] = *(const v4i*)(Wg + koff);
      b.q[1] = *(const v4i*)(Wg + koff + 8);
      ag.v = wmma_bf16(a, b, ag.v);
    }
    for (int i = 0; i < 8; ++i) {
      float g  = tanhf_(ag.v[i]);
      float hn = zv[i] * hreg[i] + (1.0f - zv[i]) * g;
      hreg[i] = hn;
      int M = mt * 16 + lh * 8 + i;
      __bf16 hb = f2bf(hn);
      hbf[M][N] = hb;
      outb[((size_t)M * GS + t) * GH + N] = hb;
      if (t == GS - 1) hout[((size_t)M * GL + layer) * GH + N] = hn;
    }
    __syncthreads();
  }
}

// ---------------------------------------------------------------------------
// Output projection: y[b,t,:] = out1[b,t,:] @ Why^T + by, f32 row-major.
// ---------------------------------------------------------------------------
__global__ __launch_bounds__(256) void out_gemm(
    const __bf16* __restrict__ Ain, const __bf16* __restrict__ Wy,
    const float* __restrict__ by, float* __restrict__ y) {
  __shared__ __bf16 As[16][GH + 8];
  const int t = blockIdx.y, mt = blockIdx.x, tid = threadIdx.x;
  {
    const int row = tid >> 4, c0 = (tid & 15) * 16;
    const __bf16* src = Ain + ((size_t)(mt * 16 + row) * GS + t) * GH + c0;
    *(v4i*)&As[row][c0]     = *(const v4i*)src;
    *(v4i*)&As[row][c0 + 8] = *(const v4i*)(src + 8);
  }
  __syncthreads();
  const int w = tid >> 5, l = tid & 31, lr = l & 15, lh = l >> 4;
  for (int jj = 0; jj < 2; ++jj) {
    const int nt = w * 2 + jj;
    const int n  = nt * 16 + lr;
    v8f acc; { float v = by[n]; for (int i = 0; i < 8; ++i) acc[i] = v; }
    for (int ks = 0; ks < 8; ++ks) {
      BF a, b;
      const __bf16* ap = &As[lr][ks * 32 + lh * 16];
      a.q[0] = *(const v4i*)ap;
      a.q[1] = *(const v4i*)(ap + 8);
      const size_t koff = (size_t)n * GH + ks * 32 + lh * 16;
      b.q[0] = *(const v4i*)(Wy + koff);
      b.q[1] = *(const v4i*)(Wy + koff + 8);
      acc = wmma_bf16(a, b, acc);
    }
    for (int i = 0; i < 8; ++i) {
      int M = mt * 16 + lh * 8 + i;
      y[((size_t)M * GS + t) * GO + n] = acc[i];
    }
  }
}

// ---------------------------------------------------------------------------
extern "C" void kernel_launch(void* const* d_in, const int* in_sizes, int n_in,
                              void* d_out, int out_size, void* d_ws, size_t ws_size,
                              hipStream_t stream) {
  (void)in_sizes; (void)n_in; (void)out_size; (void)ws_size;
  const float* x   = (const float*)d_in[0];
  const float* h0  = (const float*)d_in[1];
  const float* Wxz = (const float*)d_in[2];
  const float* Whz = (const float*)d_in[3];
  const float* bz  = (const float*)d_in[4];
  const float* Wxr = (const float*)d_in[5];
  const float* Whr = (const float*)d_in[6];
  const float* br  = (const float*)d_in[7];
  const float* Wxg = (const float*)d_in[8];
  const float* Whg = (const float*)d_in[9];
  const float* bg  = (const float*)d_in[10];
  const float* Why = (const float*)d_in[11];
  const float* by  = (const float*)d_in[12];

  float* out        = (float*)d_out;
  float* hidden_out = out + (size_t)GB * GS * GO;

  char*   ws   = (char*)d_ws;
  float*  Xp   = (float*)ws;                                       // 3*B*S*H f32
  __bf16* outb = (__bf16*)(ws + (size_t)3 * GB * GS * GH * 4);     // B*S*H bf16
  __bf16* wb   = (__bf16*)(ws + (size_t)3 * GB * GS * GH * 4
                              + (size_t)GB * GS * GH * 2);
  __bf16* wxz = wb;
  __bf16* whz = wxz + 2 * HH;
  __bf16* wxr = whz + 2 * HH;
  __bf16* whr = wxr + 2 * HH;
  __bf16* wxg = whr + 2 * HH;
  __bf16* whg = wxg + 2 * HH;
  __bf16* wy  = whg + 2 * HH;

  cvt_bf16<<<(2 * HH + 255) / 256, 256, 0, stream>>>(Wxz, wxz, 2 * HH);
  cvt_bf16<<<(2 * HH + 255) / 256, 256, 0, stream>>>(Whz, whz, 2 * HH);
  cvt_bf16<<<(2 * HH + 255) / 256, 256, 0, stream>>>(Wxr, wxr, 2 * HH);
  cvt_bf16<<<(2 * HH + 255) / 256, 256, 0, stream>>>(Whr, whr, 2 * HH);
  cvt_bf16<<<(2 * HH + 255) / 256, 256, 0, stream>>>(Wxg, wxg, 2 * HH);
  cvt_bf16<<<(2 * HH + 255) / 256, 256, 0, stream>>>(Whg, whg, 2 * HH);
  cvt_bf16<<<(HH + 255) / 256, 256, 0, stream>>>(Why, wy, HH);

  dim3 ggrid(2, GS);
  // layer 0
  xproj_gemm<<<ggrid, 256, 0, stream>>>(x, nullptr, 0, wxz, wxr, wxg,
                                        bz, br, bg, Xp);
  gru_scan<<<1, 1024, 0, stream>>>(Xp, whz, whr, whg, h0, 0, outb, hidden_out);
  // layer 1 (input = layer-0 output, already bf16)
  xproj_gemm<<<ggrid, 256, 0, stream>>>(nullptr, outb, 1, wxz + HH, wxr + HH,
                                        wxg + HH, bz + GH, br + GH, bg + GH, Xp);
  gru_scan<<<1, 1024, 0, stream>>>(Xp, whz + HH, whr + HH, whg + HH, h0, 1,
                                   outb, hidden_out);
  // output projection
  out_gemm<<<ggrid, 256, 0, stream>>>(outb, wy, by, out);
}
